// RNN_s_23596550324266
// MI455X (gfx1250) — compile-verified, matrix operand order checked by reading, not run on hardware
//
#include <hip/hip_runtime.h>
#include <stdint.h>

#define B_J0  0.01f
#define BETA  1.8f

typedef __attribute__((ext_vector_type(2)))  float  v2f;
typedef __attribute__((ext_vector_type(8)))  float  v8f;
typedef __attribute__((ext_vector_type(16))) __bf16 v16bf;
typedef __attribute__((ext_vector_type(4)))  unsigned int u32x4;
typedef __attribute__((ext_vector_type(8)))  int i32x8;
typedef __attribute__((ext_vector_type(4)))  int i32x4;

static constexpr int BSZ = 256;   // batch
static constexpr int S   = 250;   // timesteps
static constexpr int C   = 700;   // input channels
static constexpr int H   = 256;   // hidden
static constexpr int O   = 20;    // outputs
static constexpr int H2  = 512;   // concat hidden
static constexpr int KPAD = H + 8; // padded bf16 row (528B stride -> conflict-free ds_load_b128)

// TDM-padded readout slab: 512 bf16 row + 1 DWORD pad = 514 bf16 per row
static constexpr int SLABROW = H2 + 2;

// ---------- workspace layout ----------
static constexpr size_t PROJ_ELEMS = (size_t)S * BSZ * H;
static constexpr size_t OFF_PROJF  = 0;
static constexpr size_t OFF_PROJB  = PROJ_ELEMS * 4;           // bytes
static constexpr size_t OFF_SPK    = PROJ_ELEMS * 8;           // bf16 [S][B][512]

// ---------- bf16 helpers ----------
static __device__ inline __bf16 f2bf(float f) {
    union { float f; uint32_t u; } x; x.f = f;
    uint32_t r = x.u + 0x7FFFu + ((x.u >> 16) & 1u);   // RNE
    union { uint16_t u; __bf16 b; } y; y.u = (uint16_t)(r >> 16);
    return y.b;
}
static __device__ inline float bf2f(__bf16 b) {
    union { uint16_t u; __bf16 b; } y; y.b = b;
    union { uint32_t u; float f; } x; x.u = ((uint32_t)y.u) << 16;
    return x.f;
}

// 16-bit 16x32 WMMA A/B fragment from LDS (KPAD row stride), per §7.12.2.
static __device__ inline v16bf load_frag_bf16(const __bf16* mat, int row, int kb, int hi) {
    const __bf16* p = mat + (size_t)row * KPAD + kb + hi * 8;
    union { uint4 q[2]; v16bf v; } u;
    u.q[0] = *(const uint4*)(p);
    u.q[1] = *(const uint4*)(p + 16);
    return u.v;
}

// =====================================================================
// Kernel 1: proj[s,b,h] = x[b, s or S-1-s, :] @ W_in + b_in   (fp32 WMMA)
// =====================================================================
__global__ __launch_bounds__(256)
void proj_gemm_k(const float* __restrict__ x,
                 const float* __restrict__ Wf, const float* __restrict__ bf,
                 const float* __restrict__ Wb, const float* __restrict__ bb,
                 float* __restrict__ projF, float* __restrict__ projB)
{
    const int lane = threadIdx.x & 31;
    const int wid  = threadIdx.x >> 5;
    const int lh   = lane & 15;
    const int hi   = lane >> 4;
    const int cell = blockIdx.z;

    const int tileM = blockIdx.x * 64 + (wid & 3) * 16;
    const int tileN = blockIdx.y * 32 + (wid >> 2) * 16;

    const float* W  = cell ? Wb : Wf;
    const float* bi = cell ? bb : bf;
    float* out      = cell ? projB : projF;

    const int r  = tileM + lh;
    const int s  = r >> 8;
    const int b  = r & 255;
    const int ss = cell ? (S - 1 - s) : s;
    const float* xrow = x + ((size_t)b * S + ss) * C;

    const int n = tileN + lh;
    const float bias = bi[n];
    v8f acc;
    #pragma unroll
    for (int j = 0; j < 8; ++j) acc[j] = bias;

    for (int k = 0; k < C; k += 4) {
        const int k0 = k + hi * 2;
        v2f a = *(const v2f*)(xrow + k0);
        v2f bfrag;
        bfrag.x = W[(size_t)k0 * H + n];
        bfrag.y = W[(size_t)(k0 + 1) * H + n];
        acc = __builtin_amdgcn_wmma_f32_16x16x4_f32(false, a, false, bfrag,
                                                    (short)0, acc, false, false);
    }

    #pragma unroll
    for (int j = 0; j < 8; ++j) {
        const int m = tileM + j + 8 * hi;
        out[(size_t)m * H + n] = acc[j];
    }
}

// =====================================================================
// Kernel 2: 250-step adaptive-LIF recurrence; 16 WGs x 16 batch rows.
// W_rec resident in LDS (bf16, transposed). Spike GEMM: bf16 WMMA.
// Spike copy-out uses GLOBAL_STORE_ASYNC_FROM_LDS_B128 (ASYNCcnt).
// =====================================================================
__global__ __launch_bounds__(256)
void recurrence_k(const float* __restrict__ projF, const float* __restrict__ projB,
                  const float* __restrict__ WrecF, const float* __restrict__ brecF,
                  const float* __restrict__ taumF, const float* __restrict__ tauaF,
                  const float* __restrict__ WrecB, const float* __restrict__ brecB,
                  const float* __restrict__ taumB, const float* __restrict__ tauaB,
                  __bf16* __restrict__ spkOut)   // [S][B][512] bf16
{
    extern __shared__ char lds_raw[];
    __bf16* WTf  = (__bf16*)lds_raw;
    __bf16* WTb  = WTf + (size_t)H * KPAD;
    __bf16* spkF = WTb + (size_t)H * KPAD;
    __bf16* spkB = spkF + 16 * KPAD;

    const int tid  = threadIdx.x;
    const int lane = tid & 31;
    const int wid  = tid >> 5;
    const int lh   = lane & 15;
    const int hi   = lane >> 4;
    const int bslice = blockIdx.x * 16;

    for (int i = tid; i < H * H; i += 256) {
        const int k = i >> 8, n = i & 255;
        WTf[(size_t)n * KPAD + k] = f2bf(WrecF[i]);
        WTb[(size_t)n * KPAD + k] = f2bf(WrecB[i]);
    }
    for (int i = tid; i < 16 * KPAD; i += 256) { spkF[i] = f2bf(0.f); spkB[i] = f2bf(0.f); }
    __syncthreads();

    float alF[2], roF[2], brF[2], alB[2], roB[2], brB[2];
    #pragma unroll
    for (int tl = 0; tl < 2; ++tl) {
        const int n = wid * 32 + tl * 16 + lh;
        alF[tl] = __expf(-1.f / taumF[n]);  roF[tl] = __expf(-1.f / tauaF[n]);  brF[tl] = brecF[n];
        alB[tl] = __expf(-1.f / taumB[n]);  roB[tl] = __expf(-1.f / tauaB[n]);  brB[tl] = brecB[n];
    }

    float memF[2][8], bbF[2][8], memB[2][8], bbB[2][8];
    #pragma unroll
    for (int tl = 0; tl < 2; ++tl)
        #pragma unroll
        for (int j = 0; j < 8; ++j) {
            memF[tl][j] = 0.f; memB[tl][j] = 0.f;
            bbF[tl][j] = B_J0; bbB[tl][j] = B_J0;
        }

    for (int t = 0; t < S; ++t) {
        v8f dF[2], dB[2];
        #pragma unroll
        for (int tl = 0; tl < 2; ++tl) {
            #pragma unroll
            for (int j = 0; j < 8; ++j) { dF[tl][j] = brF[tl]; dB[tl][j] = brB[tl]; }
        }
        #pragma unroll
        for (int ks = 0; ks < 8; ++ks) {
            const int kb = ks * 32;
            v16bf aF = load_frag_bf16(spkF, lh, kb, hi);
            v16bf aB = load_frag_bf16(spkB, lh, kb, hi);
            #pragma unroll
            for (int tl = 0; tl < 2; ++tl) {
                const int n0 = wid * 32 + tl * 16;
                v16bf bF = load_frag_bf16(WTf + (size_t)n0 * KPAD, lh, kb, hi);
                v16bf bB = load_frag_bf16(WTb + (size_t)n0 * KPAD, lh, kb, hi);
                dF[tl] = __builtin_amdgcn_wmma_f32_16x16x32_bf16(false, aF, false, bF,
                                                                 (short)0, dF[tl], false, false);
                dB[tl] = __builtin_amdgcn_wmma_f32_16x16x32_bf16(false, aB, false, bB,
                                                                 (short)0, dB[tl], false, false);
            }
        }

        float nsF[2][8], nsB[2][8];
        #pragma unroll
        for (int tl = 0; tl < 2; ++tl) {
            const int n = wid * 32 + tl * 16 + lh;
            #pragma unroll
            for (int j = 0; j < 8; ++j) {
                const int m  = j + 8 * hi;
                const int bg = bslice + m;
                const size_t pidx = ((size_t)t * BSZ + bg) * H + n;
                {
                    const float d  = dF[tl][j] + projF[pidx];
                    const float os = bf2f(spkF[m * KPAD + n]);
                    const float nb = roF[tl] * bbF[tl][j] + (1.f - roF[tl]) * os;
                    bbF[tl][j] = nb;
                    const float thr = B_J0 + BETA * nb;
                    const float nm  = memF[tl][j] * alF[tl] + (1.f - alF[tl]) * d - thr * os;
                    memF[tl][j] = nm;
                    nsF[tl][j]  = (nm - thr) > 0.f ? 1.f : 0.f;
                }
                {
                    const float d  = dB[tl][j] + projB[pidx];
                    const float os = bf2f(spkB[m * KPAD + n]);
                    const float nb = roB[tl] * bbB[tl][j] + (1.f - roB[tl]) * os;
                    bbB[tl][j] = nb;
                    const float thr = B_J0 + BETA * nb;
                    const float nm  = memB[tl][j] * alB[tl] + (1.f - alB[tl]) * d - thr * os;
                    memB[tl][j] = nm;
                    nsB[tl][j]  = (nm - thr) > 0.f ? 1.f : 0.f;
                }
            }
        }
        __syncthreads();

        #pragma unroll
        for (int tl = 0; tl < 2; ++tl) {
            const int n = wid * 32 + tl * 16 + lh;
            #pragma unroll
            for (int j = 0; j < 8; ++j) {
                const int m = j + 8 * hi;
                spkF[m * KPAD + n] = f2bf(nsF[tl][j]);
                spkB[m * KPAD + n] = f2bf(nsB[tl][j]);
            }
        }
        __syncthreads();

        // ---- async LDS -> global spike copy (CDNA5 ASYNCcnt path) ----
        {
            const int m  = tid >> 4;     // 0..15 rows
            const int ch = tid & 15;     // 16 chunks, 16 bf16 (32B) apart
            __bf16* dst = spkOut + ((size_t)t * BSZ + (bslice + m)) * H2;
            const uint64_t gaF = (uint64_t)(uintptr_t)(dst + ch * 16);
            const uint64_t gaB = (uint64_t)(uintptr_t)(dst + H + ch * 16);
            // generic->LDS offset: addr[31:0] is the LDS byte offset (§10.2)
            const uint32_t loF = (uint32_t)(uintptr_t)(spkF + m * KPAD + ch * 16);
            const uint32_t loB = (uint32_t)(uintptr_t)(spkB + m * KPAD + ch * 16);
            asm volatile("global_store_async_from_lds_b128 %0, %1, off"
                         :: "v"(gaF), "v"(loF) : "memory");
            asm volatile("global_store_async_from_lds_b128 %0, %1, off"
                         :: "v"(gaB), "v"(loB) : "memory");
            // drain before the LDS spike buffers are overwritten next step
            asm volatile("s_wait_asynccnt 0" ::: "memory");
        }
    }
}

// =====================================================================
// Kernel 3: readout integrator + log_softmax. Per timestep the 16x512
// bf16 spike slab is DMA'd into LDS by the Tensor Data Mover with a
// 2D D# (pad_interval=256 DWORDs, pad_amount=1 DWORD -> bank-rotated
// rows), synced with s_wait_tensorcnt, then fed to fp32 WMMA.
// =====================================================================
__global__ __launch_bounds__(32)
void readout_k(const __bf16* __restrict__ spk,   // [S][B][512]
               const float* __restrict__ Wout,   // [512][20]
               const float* __restrict__ bout,   // [20]
               const float* __restrict__ tauo,   // [20]
               float* __restrict__ out)          // [256][20]
{
    __shared__ float  Wl[H2 * O];          // 40KB
    __shared__ float  ml[16][32];
    __shared__ __bf16 slab[16 * SLABROW];  // TDM destination, 16,448B

    const int lane  = threadIdx.x;
    const int lh    = lane & 15;
    const int hi    = lane >> 4;
    const int bbase = blockIdx.x * 16;

    for (int i = lane; i < H2 * O; i += 32) Wl[i] = Wout[i];
    __syncthreads();

    const uint32_t slab_off = (uint32_t)(uintptr_t)(void*)slab;  // LDS byte offset

    float alpha[2], bo[2], mo[2][8];
    bool valid[2];
    #pragma unroll
    for (int tl = 0; tl < 2; ++tl) {
        const int n = tl * 16 + lh;
        valid[tl] = (n < O);
        alpha[tl] = valid[tl] ? __expf(-1.f / tauo[n]) : 0.f;
        bo[tl]    = valid[tl] ? bout[n] : 0.f;
        #pragma unroll
        for (int j = 0; j < 8; ++j) mo[tl][j] = 0.f;
    }

    for (int t = 0; t < S; ++t) {
        // ---- TDM: DMA 16x512 bf16 tile (row-padded) into LDS ----
        {
            const uint64_t ga = (uint64_t)(uintptr_t)(spk + ((size_t)t * BSZ + bbase) * H2);
            u32x4 g0;
            g0[0] = 1u;                                   // count=1, user descriptor
            g0[1] = slab_off;                             // lds_addr
            g0[2] = (uint32_t)ga;                         // global_addr[31:0]
            g0[3] = (uint32_t)((ga >> 32) & 0x01FFFFFFu)  // global_addr[56:32]
                    | (2u << 30);                         // type=2 (image)
            i32x8 g1;
            g1[0] = (int)((1u << 16)      // data_size = 2B
                          | (1u << 20)    // pad_enable
                          | (7u << 22));  // pad_interval = 256 DWORDs (one row)
                                          // pad_amount = 0 -> 1 DWORD
            g1[1] = (int)(512u << 16);    // tensor_dim0 = 512
            g1[2] = (int)(16u  << 16);    // tensor_dim1 = 16
            g1[3] = (int)(512u << 16);    // tile_dim0   = 512
            g1[4] = (int)(16u);           // tile_dim1   = 16
            g1[5] = (int)(512);           // tensor_dim0_stride = 512
            g1[6] = 0;
            g1[7] = 0;
            i32x4 z4 = {0, 0, 0, 0};
            i32x8 z8 = {0, 0, 0, 0, 0, 0, 0, 0};
            // amdgpu-toolchain (clang-23) 6-arg form: (g0, g1, g2, g3, g4, cpol)
            __builtin_amdgcn_tensor_load_to_lds(g0, g1, z4, z4, z8, 0);
            __builtin_amdgcn_s_wait_tensorcnt(0);
        }

        const __bf16* srow = slab + lh * SLABROW;   // this lane's batch row
        v8f acc[2];
        #pragma unroll
        for (int tl = 0; tl < 2; ++tl)
            #pragma unroll
            for (int j = 0; j < 8; ++j) acc[tl][j] = bo[tl];

        for (int k = 0; k < H2; k += 4) {
            const int k0 = k + hi * 2;
            v2f a;
            a.x = bf2f(srow[k0]);
            a.y = bf2f(srow[k0 + 1]);
            #pragma unroll
            for (int tl = 0; tl < 2; ++tl) {
                const int n = tl * 16 + lh;
                v2f bfr;
                bfr.x = valid[tl] ? Wl[k0 * O + n] : 0.f;
                bfr.y = valid[tl] ? Wl[(k0 + 1) * O + n] : 0.f;
                acc[tl] = __builtin_amdgcn_wmma_f32_16x16x4_f32(false, a, false, bfr,
                                                                (short)0, acc[tl], false, false);
            }
        }
        #pragma unroll
        for (int tl = 0; tl < 2; ++tl)
            #pragma unroll
            for (int j = 0; j < 8; ++j)
                mo[tl][j] = mo[tl][j] * alpha[tl] + (1.f - alpha[tl]) * acc[tl][j];
    }

    #pragma unroll
    for (int tl = 0; tl < 2; ++tl)
        #pragma unroll
        for (int j = 0; j < 8; ++j)
            ml[j + 8 * hi][tl * 16 + lh] = mo[tl][j];
    __syncthreads();

    if (lane < 16) {
        const int m = lane;
        float mx = -1e30f;
        for (int o = 0; o < O; ++o) mx = fmaxf(mx, ml[m][o]);
        float se = 0.f;
        for (int o = 0; o < O; ++o) se += __expf(ml[m][o] - mx);
        const float lse = mx + __logf(se);
        for (int o = 0; o < O; ++o)
            out[(size_t)(bbase + m) * O + o] = ml[m][o] - lse;
    }
}

// =====================================================================
extern "C" void kernel_launch(void* const* d_in, const int* in_sizes, int n_in,
                              void* d_out, int out_size, void* d_ws, size_t ws_size,
                              hipStream_t stream) {
    (void)in_sizes; (void)n_in; (void)out_size; (void)ws_size;
    const float* x       = (const float*)d_in[0];
    const float* W_in_f  = (const float*)d_in[1];
    const float* b_in_f  = (const float*)d_in[2];
    const float* W_rec_f = (const float*)d_in[3];
    const float* b_rec_f = (const float*)d_in[4];
    const float* tau_m_f = (const float*)d_in[5];
    const float* tau_a_f = (const float*)d_in[6];
    const float* W_in_b  = (const float*)d_in[7];
    const float* b_in_b  = (const float*)d_in[8];
    const float* W_rec_b = (const float*)d_in[9];
    const float* b_rec_b = (const float*)d_in[10];
    const float* tau_m_b = (const float*)d_in[11];
    const float* tau_a_b = (const float*)d_in[12];
    const float* W_out   = (const float*)d_in[13];
    const float* b_out   = (const float*)d_in[14];
    const float* tau_o   = (const float*)d_in[15];

    char* ws = (char*)d_ws;
    float*  projF = (float*)(ws + OFF_PROJF);
    float*  projB = (float*)(ws + OFF_PROJB);
    __bf16* spks  = (__bf16*)(ws + OFF_SPK);

    proj_gemm_k<<<dim3((S * BSZ) / 64, H / 32, 2), 256, 0, stream>>>(
        x, W_in_f, b_in_f, W_in_b, b_in_b, projF, projB);

    const size_t ldsBytes = (size_t)(2 * H + 32) * KPAD * sizeof(__bf16); // 287,232 B
    recurrence_k<<<dim3(BSZ / 16), 256, ldsBytes, stream>>>(
        projF, projB, W_rec_f, b_rec_f, tau_m_f, tau_a_f,
        W_rec_b, b_rec_b, tau_m_b, tau_a_b, spks);

    readout_k<<<dim3(BSZ / 16), 32, 0, stream>>>(spks, W_out, b_out, tau_o, (float*)d_out);
}
